// DeformablePointAttention_19911468384449
// MI455X (gfx1250) — matrix operand order repaired and checked by reading, not compiled, not run on hardware
//
#include <hip/hip_runtime.h>
#include <math.h>

// ---------------------------------------------------------------------------
// DeformablePointAttention for MI455X (gfx1250), wave32 + WMMA bf16.
// Fixed problem size from the reference: B=2, N=1024, DIM=256, H=8, P=16, K3=3
// ---------------------------------------------------------------------------

typedef __bf16 bf16;
typedef __attribute__((ext_vector_type(16))) __bf16 v16bf;
typedef __attribute__((ext_vector_type(8)))  __bf16 v8bf;
typedef __attribute__((ext_vector_type(8)))  float  v8f;

union BF16x16 { v16bf v; v8bf h[2]; };

constexpr int BATCH = 2;
constexpr int NPTS  = 1024;
constexpr int DIMC  = 256;
constexpr int HEADS = 8;
constexpr int PPTS  = 16;
constexpr int DH    = 32;                 // DIM / H
constexpr int M_TOK = BATCH * NPTS;       // 2048 token rows
constexpr int KA0   = 288;                // 259 (x_t ++ xyz) padded to mult of 32
constexpr int NSAMP = M_TOK * HEADS * PPTS; // 262144 sample points

__device__ __forceinline__ float gelu_f(float x) {
    return 0.5f * x * (1.0f + erff(x * 0.70710678118654752440f));
}

// ---------------------------------------------------------------------------
// WMMA GEMM:  C[M x Nout] = A[M x Kpad](bf16, row major)
//                         @ Wt[Nout x Kpad](bf16, W transposed, K zero-padded)
// epi: 0 = +bias, 1 = gelu(+bias).  Cf (f32) and/or Cb (bf16) outputs.
// One wave -> 16x64 tile strip (4 accumulators sharing one A fragment).
// Software-pipelined: iteration i's WMMAs consume fragments loaded at i-1,
// so the K-step's 10 b128 loads stay in flight under the matrix pipe.
// Requires Nout % 64 == 0 and Kpad >= 64 (true for all GEMMs here).
// ---------------------------------------------------------------------------
__global__ __launch_bounds__(256) void gemm_wmma(
    const bf16* __restrict__ A, const bf16* __restrict__ Wt,
    const float* __restrict__ bias, float* __restrict__ Cf,
    bf16* __restrict__ Cb, int M, int Nout, int Kpad, int epi)
{
    const int lane    = threadIdx.x & 31;
    const int wave    = threadIdx.x >> 5;
    const int tilesN4 = Nout >> 6;                        // groups of 4 N-tiles
    const int grp     = blockIdx.x * (blockDim.x >> 5) + wave;
    const int tm      = grp / tilesN4;
    const int tg      = grp - tm * tilesN4;
    if (tm * 16 >= M) return;                 // wave-uniform exit (EXEC stays full)

    const int half = lane >> 4;
    const int l16  = lane & 15;
    const int row  = tm * 16 + l16;           // A-fragment row for this lane
    const int col0 = tg * 64 + l16;           // first C column for this lane

    const bf16*  ap  = A  + (size_t)row  * Kpad + half * 8;   // K = k0+half*8+{0..7}, +16
    const bf16*  bp  = Wt + (size_t)col0 * Kpad + half * 8;   // K = k0+half*8+{0..7}, +16
    const size_t bstr = (size_t)16 * Kpad;                    // next N-tile of Wt

    v8f acc[4] = {{}, {}, {}, {}};

    // ---- prologue: load K-step 0 ----
    BF16x16 a, b[4];
    a.h[0] = *(const v8bf*)(ap);
    a.h[1] = *(const v8bf*)(ap + 16);
#pragma unroll
    for (int j = 0; j < 4; ++j) {
        b[j].h[0] = *(const v8bf*)(bp + (size_t)j * bstr);
        b[j].h[1] = *(const v8bf*)(bp + (size_t)j * bstr + 16);
    }

    // ---- pipelined body: load k0 while multiplying k0-32 ----
    for (int k0 = 32; k0 < Kpad; k0 += 32) {
        BF16x16 an, bn[4];
        an.h[0] = *(const v8bf*)(ap + k0);
        an.h[1] = *(const v8bf*)(ap + k0 + 16);
#pragma unroll
        for (int j = 0; j < 4; ++j) {
            bn[j].h[0] = *(const v8bf*)(bp + (size_t)j * bstr + k0);
            bn[j].h[1] = *(const v8bf*)(bp + (size_t)j * bstr + k0 + 16);
        }
#pragma unroll
        for (int j = 0; j < 4; ++j) {
            acc[j] = __builtin_amdgcn_wmma_f32_16x16x32_bf16(
                false, a.v, false, b[j].v, (short)0, acc[j], false, false);
        }
        a = an;
#pragma unroll
        for (int j = 0; j < 4; ++j) b[j] = bn[j];
    }

    // ---- epilogue: last K-step ----
#pragma unroll
    for (int j = 0; j < 4; ++j) {
        acc[j] = __builtin_amdgcn_wmma_f32_16x16x32_bf16(
            false, a.v, false, b[j].v, (short)0, acc[j], false, false);
    }

#pragma unroll
    for (int j = 0; j < 4; ++j) {
        const int   col = col0 + j * 16;
        const float bv  = bias ? bias[col] : 0.0f;
#pragma unroll
        for (int r = 0; r < 8; ++r) {
            const int orow = tm * 16 + half * 8 + r;   // C/D layout: VGPR r -> M = r + 8*half
            float v = acc[j][r] + bv;
            if (epi == 1) v = gelu_f(v);
            const size_t o = (size_t)orow * Nout + col;
            if (Cf) Cf[o] = v;
            if (Cb) Cb[o] = (bf16)v;
        }
    }
}

// Wt[n*Kpad + k] = (k < K) ? bf16(W[k*Nout + n]) : 0   (transpose + cast + pad)
__global__ void cast_weight_t(const float* __restrict__ W, bf16* __restrict__ Wt,
                              int K, int Nout, int Kpad)
{
    const int i = blockIdx.x * blockDim.x + threadIdx.x;
    if (i >= Nout * Kpad) return;
    const int n = i / Kpad;
    const int k = i - n * Kpad;
    const float v = (k < K) ? W[(size_t)k * Nout + n] : 0.0f;
    Wt[i] = (bf16)v;
}

// xt[m][c] = x[b][c][n] (f32 identity) ; A0[m][0..255]=x_t, [256..258]=xyz, pad 0
__global__ void build_a0(const float* __restrict__ x, const float* __restrict__ xyz,
                         float* __restrict__ xt, bf16* __restrict__ A0)
{
    const int i = blockIdx.x * blockDim.x + threadIdx.x;
    if (i >= M_TOK * KA0) return;
    const int m = i / KA0;
    const int c = i - m * KA0;
    const int b = m >> 10;            // / NPTS
    const int n = m & (NPTS - 1);
    float v;
    if (c < DIMC) {
        v = x[((size_t)b * DIMC + c) * NPTS + n];
        xt[(size_t)m * DIMC + c] = v;
    } else if (c < DIMC + 3) {
        v = xyz[((size_t)b * NPTS + n) * 3 + (c - DIMC)];
    } else {
        v = 0.0f;
    }
    A0[i] = (bf16)v;
}

// offsets <- tanh(offsets) * |offset_scale|   (in place, M_TOK*384 elems)
__global__ void tanh_scale(float* __restrict__ off, const float* __restrict__ sc, int total)
{
    const int i = blockIdx.x * blockDim.x + threadIdx.x;
    if (i < total) off[i] = tanhf(off[i]) * fabsf(sc[0]);
}

// LayerNorm over 256-wide rows, optional residual add and gelu; one wave/row.
__global__ __launch_bounds__(256) void ln_rows(
    const float* __restrict__ in, const float* __restrict__ resid,
    const float* __restrict__ g, const float* __restrict__ bta,
    float* __restrict__ outf, bf16* __restrict__ outb, int do_gelu)
{
    const int lane = threadIdx.x & 31;
    const int wave = threadIdx.x >> 5;
    const int row  = blockIdx.x * 8 + wave;
    if (row >= M_TOK) return;
    const float* ip = in + (size_t)row * DIMC;
    const float* rp = resid ? resid + (size_t)row * DIMC : nullptr;

    float v[8]; float s = 0.0f;
#pragma unroll
    for (int j = 0; j < 8; ++j) {
        const int c = lane + j * 32;
        float t = ip[c] + (rp ? rp[c] : 0.0f);
        v[j] = t; s += t;
    }
#pragma unroll
    for (int o = 16; o > 0; o >>= 1) s += __shfl_xor(s, o, 32);
    const float mean = s * (1.0f / DIMC);
    float q = 0.0f;
#pragma unroll
    for (int j = 0; j < 8; ++j) { const float d = v[j] - mean; q += d * d; }
#pragma unroll
    for (int o = 16; o > 0; o >>= 1) q += __shfl_xor(q, o, 32);
    const float rstd = rsqrtf(q * (1.0f / DIMC) + 1e-5f);
#pragma unroll
    for (int j = 0; j < 8; ++j) {
        const int c = lane + j * 32;
        float t = (v[j] - mean) * rstd * g[c] + bta[c];
        if (do_gelu) t = gelu_f(t);
        const size_t o = (size_t)row * DIMC + c;
        if (outf) outf[o] = t;
        if (outb) outb[o] = (bf16)t;
    }
}

// Final LN with residual, output transposed to (B, DIM, N) f32.
__global__ __launch_bounds__(256) void ln_final(
    const float* __restrict__ f2, const float* __restrict__ ln1,
    const float* __restrict__ g, const float* __restrict__ bta,
    float* __restrict__ out)
{
    const int lane = threadIdx.x & 31;
    const int wave = threadIdx.x >> 5;
    const int row  = blockIdx.x * 8 + wave;
    if (row >= M_TOK) return;
    const int b = row >> 10, n = row & (NPTS - 1);
    const float* ip = f2  + (size_t)row * DIMC;
    const float* rp = ln1 + (size_t)row * DIMC;

    float v[8]; float s = 0.0f;
#pragma unroll
    for (int j = 0; j < 8; ++j) {
        const int c = lane + j * 32;
        float t = ip[c] + rp[c];
        v[j] = t; s += t;
    }
#pragma unroll
    for (int o = 16; o > 0; o >>= 1) s += __shfl_xor(s, o, 32);
    const float mean = s * (1.0f / DIMC);
    float q = 0.0f;
#pragma unroll
    for (int j = 0; j < 8; ++j) { const float d = v[j] - mean; q += d * d; }
#pragma unroll
    for (int o = 16; o > 0; o >>= 1) q += __shfl_xor(q, o, 32);
    const float rstd = rsqrtf(q * (1.0f / DIMC) + 1e-5f);
#pragma unroll
    for (int j = 0; j < 8; ++j) {
        const int c = lane + j * 32;
        const float t = (v[j] - mean) * rstd * g[c] + bta[c];
        out[((size_t)b * DIMC + c) * NPTS + n] = t;   // (B, DIM, N)
    }
}

// Top-3 nearest xyz neighbors per sampled point + inverse-distance weights.
// One thread per sample; batch xyz staged in LDS (12 KB).
__global__ __launch_bounds__(256) void knn_topk(
    const float* __restrict__ xyz, const float* __restrict__ off,
    int* __restrict__ idx, float* __restrict__ wgt)
{
    __shared__ float sx[NPTS], sy[NPTS], sz[NPTS];
    const int s = blockIdx.x * 256 + threadIdx.x;     // 262144 samples, 1024 blocks
    const int b = s / (NPTS * HEADS * PPTS);          // whole block in one batch
    const float* xb = xyz + (size_t)b * NPTS * 3;
    for (int t = threadIdx.x; t < NPTS; t += 256) {
        sx[t] = xb[t * 3 + 0]; sy[t] = xb[t * 3 + 1]; sz[t] = xb[t * 3 + 2];
    }
    __syncthreads();

    const int r = s - b * NPTS * HEADS * PPTS;
    const int n = r / (HEADS * PPTS);
    const float px = xb[n * 3 + 0] + off[(size_t)s * 3 + 0];
    const float py = xb[n * 3 + 1] + off[(size_t)s * 3 + 1];
    const float pz = xb[n * 3 + 2] + off[(size_t)s * 3 + 2];

    float d0 = 3.0e38f, d1 = 3.0e38f, d2 = 3.0e38f;
    int   i0 = 0, i1 = 0, i2 = 0;
    for (int m = 0; m < NPTS; ++m) {
        const float dx = px - sx[m], dy = py - sy[m], dz = pz - sz[m];
        const float d  = dx * dx + dy * dy + dz * dz;
        if (d < d0)      { d2 = d1; i2 = i1; d1 = d0; i1 = i0; d0 = d; i0 = m; }
        else if (d < d1) { d2 = d1; i2 = i1; d1 = d;  i1 = m; }
        else if (d < d2) { d2 = d;  i2 = m; }
    }
    float w0 = 1.0f / (d0 + 1e-8f);
    float w1 = 1.0f / (d1 + 1e-8f);
    float w2 = 1.0f / (d2 + 1e-8f);
    const float ws = w0 + w1 + w2;
    w0 /= ws; w1 /= ws; w2 /= ws;
    idx[(size_t)s * 3 + 0] = i0; idx[(size_t)s * 3 + 1] = i1; idx[(size_t)s * 3 + 2] = i2;
    wgt[(size_t)s * 3 + 0] = w0; wgt[(size_t)s * 3 + 1] = w1; wgt[(size_t)s * 3 + 2] = w2;
}

// rbias = (gelu(LN64(rel @ rw0 + rb0)) @ rw1 + rb1); rel = -scaled_offsets
__global__ __launch_bounds__(256) void rbias_kernel(
    const float* __restrict__ off, const float* __restrict__ rw0,
    const float* __restrict__ rb0, const float* __restrict__ rg,
    const float* __restrict__ rbeta, const float* __restrict__ rw1,
    const float* __restrict__ rb1, float* __restrict__ rbias)
{
    const int s = blockIdx.x * blockDim.x + threadIdx.x;
    if (s >= NSAMP) return;
    const float r0 = -off[(size_t)s * 3 + 0];
    const float r1 = -off[(size_t)s * 3 + 1];
    const float r2 = -off[(size_t)s * 3 + 2];

    float hb[64];
    float sum = 0.0f;
#pragma unroll
    for (int j = 0; j < 64; ++j) {
        const float h = r0 * rw0[j] + r1 * rw0[64 + j] + r2 * rw0[128 + j] + rb0[j];
        hb[j] = h; sum += h;
    }
    const float mean = sum * (1.0f / 64.0f);
    float q = 0.0f;
#pragma unroll
    for (int j = 0; j < 64; ++j) { const float d = hb[j] - mean; q += d * d; }
    const float rstd = rsqrtf(q * (1.0f / 64.0f) + 1e-5f);
    float o = rb1[0];
#pragma unroll
    for (int j = 0; j < 64; ++j)
        o += gelu_f((hb[j] - mean) * rstd * rg[j] + rbeta[j]) * rw1[j];
    rbias[s] = o;
}

// Fused: gather K/V at top-3, IDW-interp, Q·sK + rbias, softmax over P=16,
// aw-weighted sum of sV.  One wave per (token,head); sV tile staged in LDS.
__global__ __launch_bounds__(256) void attn_kernel(
    const float* __restrict__ Q, const float* __restrict__ Kb,
    const float* __restrict__ Vb, const int* __restrict__ idx,
    const float* __restrict__ wgt, const float* __restrict__ rbias,
    bf16* __restrict__ outb)
{
    __shared__ float sV[8][PPTS][DH];   // 16 KB
    __shared__ float sAW[8][PPTS];
    const int lane = threadIdx.x & 31;
    const int wave = threadIdx.x >> 5;
    const int g    = blockIdx.x * 8 + wave;   // (m,h), 16384 total, grid exact
    const int m = g >> 3;
    const int h = g & (HEADS - 1);
    const int b = m >> 10;

    float attn = -3.0e38f;
    if (lane < PPTS) {
        const int p = lane;
        const int s = (m * HEADS + h) * PPTS + p;
        const int i0 = idx[(size_t)s * 3 + 0];
        const int i1 = idx[(size_t)s * 3 + 1];
        const int i2 = idx[(size_t)s * 3 + 2];
        const float w0 = wgt[(size_t)s * 3 + 0];
        const float w1 = wgt[(size_t)s * 3 + 1];
        const float w2 = wgt[(size_t)s * 3 + 2];
        const size_t hb = (size_t)h * DH;
        const float* K0 = Kb + ((size_t)b * NPTS + i0) * DIMC + hb;
        const float* K1 = Kb + ((size_t)b * NPTS + i1) * DIMC + hb;
        const float* K2 = Kb + ((size_t)b * NPTS + i2) * DIMC + hb;
        const float* V0 = Vb + ((size_t)b * NPTS + i0) * DIMC + hb;
        const float* V1 = Vb + ((size_t)b * NPTS + i1) * DIMC + hb;
        const float* V2 = Vb + ((size_t)b * NPTS + i2) * DIMC + hb;
        const float* qp = Q + (size_t)m * DIMC + hb;
        float acc = 0.0f;
#pragma unroll
        for (int d = 0; d < DH; ++d) {
            const float sk = w0 * K0[d] + w1 * K1[d] + w2 * K2[d];
            const float sv = w0 * V0[d] + w1 * V1[d] + w2 * V2[d];
            acc += qp[d] * sk;
            sV[wave][p][d] = sv;
        }
        attn = acc * 0.17677669529663687f + rbias[s];   // scale = 32^-0.5
    }
    // softmax across the 16 p-lanes
    float mx = attn;
#pragma unroll
    for (int o = 8; o > 0; o >>= 1) mx = fmaxf(mx, __shfl_xor(mx, o, 16));
    float e = (lane < PPTS) ? expf(attn - mx) : 0.0f;
    float se = e;
#pragma unroll
    for (int o = 8; o > 0; o >>= 1) se += __shfl_xor(se, o, 16);
    if (lane < PPTS) sAW[wave][lane] = e / se;
    __syncthreads();

    const int d = lane;
    float o = 0.0f;
#pragma unroll
    for (int p = 0; p < PPTS; ++p) o += sAW[wave][p] * sV[wave][p][d];
    outb[(size_t)m * DIMC + (size_t)h * DH + d] = (bf16)o;
}

// ---------------------------------------------------------------------------
extern "C" void kernel_launch(void* const* d_in, const int* in_sizes, int n_in,
                              void* d_out, int out_size, void* d_ws, size_t ws_size,
                              hipStream_t stream)
{
    (void)in_sizes; (void)n_in; (void)out_size; (void)ws_size;
    const float* x    = (const float*)d_in[0];
    const float* xyz  = (const float*)d_in[1];
    const float* osc  = (const float*)d_in[2];
    const float* ow0  = (const float*)d_in[3];
    const float* ob0  = (const float*)d_in[4];
    const float* og   = (const float*)d_in[5];
    const float* obet = (const float*)d_in[6];
    const float* ow1  = (const float*)d_in[7];
    const float* ob1  = (const float*)d_in[8];
    const float* ow2  = (const float*)d_in[9];
    const float* ob2  = (const float*)d_in[10];
    const float* qw   = (const float*)d_in[11];
    const float* kw   = (const float*)d_in[12];
    const float* vw   = (const float*)d_in[13];
    const float* rw0  = (const float*)d_in[14];
    const float* rb0  = (const float*)d_in[15];
    const float* rg   = (const float*)d_in[16];
    const float* rbet = (const float*)d_in[17];
    const float* rw1  = (const float*)d_in[18];
    const float* rb1  = (const float*)d_in[19];
    const float* pw   = (const float*)d_in[20];
    const float* fw1  = (const float*)d_in[21];
    const float* fb1  = (const float*)d_in[22];
    const float* fw2  = (const float*)d_in[23];
    const float* fb2  = (const float*)d_in[24];
    const float* n1g  = (const float*)d_in[25];
    const float* n1b  = (const float*)d_in[26];
    const float* n2g  = (const float*)d_in[27];
    const float* n2b  = (const float*)d_in[28];
    float* out = (float*)d_out;

    // -------- workspace carve-out (256B aligned slabs) --------
    char* base = (char*)d_ws;
    size_t woff = 0;
    auto alloc = [&](size_t bytes) -> void* {
        void* r = base + woff;
        woff += (bytes + 255) & ~(size_t)255;
        return r;
    };
    float* xt    = (float*)alloc((size_t)M_TOK * DIMC * 4);
    bf16*  A0    = (bf16*) alloc((size_t)M_TOK * KA0  * 2);
    bf16*  ow0t  = (bf16*) alloc((size_t)256  * 288  * 2);
    bf16*  ow1t  = (bf16*) alloc((size_t)128  * 256  * 2);
    bf16*  ow2t  = (bf16*) alloc((size_t)384  * 128  * 2);
    bf16*  qwt   = (bf16*) alloc((size_t)256  * 288  * 2);
    bf16*  kwt   = (bf16*) alloc((size_t)256  * 288  * 2);
    bf16*  vwt   = (bf16*) alloc((size_t)256  * 288  * 2);
    bf16*  pwt   = (bf16*) alloc((size_t)256  * 256  * 2);
    bf16*  fw1t  = (bf16*) alloc((size_t)1024 * 256  * 2);
    bf16*  fw2t  = (bf16*) alloc((size_t)256  * 1024 * 2);
    float* h0f   = (float*)alloc((size_t)M_TOK * 256 * 4);
    bf16*  h0b   = (bf16*) alloc((size_t)M_TOK * 256 * 2);
    bf16*  h1b   = (bf16*) alloc((size_t)M_TOK * 128 * 2);
    float* offs  = (float*)alloc((size_t)M_TOK * 384 * 4);
    float* Qf    = (float*)alloc((size_t)M_TOK * 256 * 4);
    float* Kf    = (float*)alloc((size_t)M_TOK * 256 * 4);
    float* Vf    = (float*)alloc((size_t)M_TOK * 256 * 4);
    int*   idx   = (int*)  alloc((size_t)NSAMP * 3 * 4);
    float* wgt   = (float*)alloc((size_t)NSAMP * 3 * 4);
    float* rbia  = (float*)alloc((size_t)NSAMP * 4);
    bf16*  attnb = (bf16*) alloc((size_t)M_TOK * 256 * 2);
    float* o1f   = (float*)alloc((size_t)M_TOK * 256 * 4);
    float* ln1f  = (float*)alloc((size_t)M_TOK * 256 * 4);
    bf16*  ln1b  = (bf16*) alloc((size_t)M_TOK * 256 * 2);
    bf16*  f1b   = (bf16*) alloc((size_t)M_TOK * 1024 * 2);
    float* f2f   = (float*)alloc((size_t)M_TOK * 256 * 4);

    auto eltGrid = [](int n) { return dim3((n + 255) / 256); };
    auto gemm = [&](const bf16* A, const bf16* Wt, const float* bias,
                    float* Cf, bf16* Cb, int M, int Nout, int Kpad, int epi) {
        const int groups = ((M + 15) / 16) * (Nout / 64);   // 4 N-tiles per wave
        const int blocks = (groups + 7) / 8;
        gemm_wmma<<<dim3(blocks), dim3(256), 0, stream>>>(A, Wt, bias, Cf, Cb,
                                                          M, Nout, Kpad, epi);
    };

    // 1) x_t (f32 identity) + bf16 A0 = [x_t | xyz | 0-pad]
    build_a0<<<eltGrid(M_TOK * KA0), dim3(256), 0, stream>>>(x, xyz, xt, A0);

    // 2) weight transpose+cast+pad
    cast_weight_t<<<eltGrid(256 * 288),  dim3(256), 0, stream>>>(ow0, ow0t, 259, 256, 288);
    cast_weight_t<<<eltGrid(128 * 256),  dim3(256), 0, stream>>>(ow1, ow1t, 256, 128, 256);
    cast_weight_t<<<eltGrid(384 * 128),  dim3(256), 0, stream>>>(ow2, ow2t, 128, 384, 128);
    cast_weight_t<<<eltGrid(256 * 288),  dim3(256), 0, stream>>>(qw,  qwt,  256, 256, 288);
    cast_weight_t<<<eltGrid(256 * 288),  dim3(256), 0, stream>>>(kw,  kwt,  256, 256, 288);
    cast_weight_t<<<eltGrid(256 * 288),  dim3(256), 0, stream>>>(vw,  vwt,  256, 256, 288);
    cast_weight_t<<<eltGrid(256 * 256),  dim3(256), 0, stream>>>(pw,  pwt,  256, 256, 256);
    cast_weight_t<<<eltGrid(1024 * 256), dim3(256), 0, stream>>>(fw1, fw1t, 256, 1024, 256);
    cast_weight_t<<<eltGrid(256 * 1024), dim3(256), 0, stream>>>(fw2, fw2t, 1024, 256, 1024);

    // 3) offset MLP: h0 -> LN+gelu -> h1(gelu) -> raw offsets -> tanh*|s|
    gemm(A0,  ow0t, ob0, h0f, nullptr, M_TOK, 256, 288, 0);
    ln_rows<<<dim3(M_TOK / 8), dim3(256), 0, stream>>>(h0f, nullptr, og, obet,
                                                       nullptr, h0b, 1);
    gemm(h0b, ow1t, ob1, nullptr, h1b, M_TOK, 128, 256, 1);
    gemm(h1b, ow2t, ob2, offs, nullptr, M_TOK, 384, 128, 0);
    tanh_scale<<<eltGrid(M_TOK * 384), dim3(256), 0, stream>>>(offs, osc, M_TOK * 384);

    // 4) Q/K/V projections (reuse padded A0; weight rows 256..287 are zero)
    gemm(A0, qwt, nullptr, Qf, nullptr, M_TOK, 256, 288, 0);
    gemm(A0, kwt, nullptr, Kf, nullptr, M_TOK, 256, 288, 0);
    gemm(A0, vwt, nullptr, Vf, nullptr, M_TOK, 256, 288, 0);

    // 5) KNN top-3 + weights, relative-position bias
    knn_topk<<<dim3(NSAMP / 256), dim3(256), 0, stream>>>(xyz, offs, idx, wgt);
    rbias_kernel<<<eltGrid(NSAMP), dim3(256), 0, stream>>>(offs, rw0, rb0, rg, rbet,
                                                           rw1, rb1, rbia);

    // 6) fused gather/interp/attention/softmax -> bf16 (input of pw GEMM)
    attn_kernel<<<dim3(M_TOK * HEADS / 8), dim3(256), 0, stream>>>(
        Qf, Kf, Vf, idx, wgt, rbia, attnb);

    // 7) output proj + LN1(residual) + FFN + LN2 (transposed store)
    gemm(attnb, pwt, nullptr, o1f, nullptr, M_TOK, 256, 256, 0);
    ln_rows<<<dim3(M_TOK / 8), dim3(256), 0, stream>>>(o1f, xt, n1g, n1b, ln1f, ln1b, 0);
    gemm(ln1b, fw1t, fb1, nullptr, f1b, M_TOK, 1024, 256, 1);
    gemm(f1b,  fw2t, fb2, f2f, nullptr, M_TOK, 256, 1024, 0);
    ln_final<<<dim3(M_TOK / 8), dim3(256), 0, stream>>>(f2f, ln1f, n2g, n2b, out);
}
